// lstm_mlp_66597762892574
// MI455X (gfx1250) — matrix operand order repaired and checked by reading, not compiled, not run on hardware
//
#include <hip/hip_runtime.h>
#include <hip/hip_bf16.h>
#include <math.h>

typedef __attribute__((ext_vector_type(16))) _Float16 v16h;
typedef __attribute__((ext_vector_type(8)))  float    v8f;

// ---------------------------------------------------------------------------
// WMMA helpers (CDNA5 gfx1250, wave32). D = A(16x32 f16) * B(32x16 f16) + C.
// All operands live in memory in *fragment-major* layout: per 16x32 (A) or
// 32x16 (B) tile, lane l owns 16 contiguous f16 (32 bytes) at [tile][l][16].
// ---------------------------------------------------------------------------
__device__ inline v8f vzero() {
  v8f v;
#pragma unroll
  for (int i = 0; i < 8; ++i) v[i] = 0.f;
  return v;
}
__device__ inline v16h hzero() {
  v16h v;
#pragma unroll
  for (int i = 0; i < 16; ++i) v[i] = (_Float16)0.f;
  return v;
}
__device__ inline v8f wmma16(v16h a, v16h b, v8f c) {
  return __builtin_amdgcn_wmma_f32_16x16x32_f16(false, a, false, b, (short)0, c,
                                                false, false);
}

// ISA A-map (16-bit A 16x32, §7.12.2): lane element e -> K offset
__device__ __host__ inline int amap_k(int lane, int e) {
  int r = e >> 1;
  return ((r < 4) ? (2 * r) : (8 + 2 * r)) + (e & 1) + ((lane >> 4) ? 8 : 0);
}

// Fast activations: single v_exp_f32 (TRANS co-executes with XDL WMMA).
__device__ inline float sigf(float x) { return 1.f / (1.f + __expf(-x)); }
__device__ inline float tanh_fast(float x) {
  x = fminf(fmaxf(x, -15.f), 15.f);
  float e = __expf(2.f * x);
  return (e - 1.f) / (e + 1.f);
}

// ---------------------------------------------------------------------------
// Conv1d (kernel 4, CIN=1, COUT=16, SAME: pad_lo=1) -> linear f16 (B,T,16)
// ---------------------------------------------------------------------------
__global__ void conv_k(const float* __restrict__ X, const float* __restrict__ w,
                       const float* __restrict__ bias, _Float16* __restrict__ out,
                       int T, int total) {
  int i = blockIdx.x * blockDim.x + threadIdx.x;
  if (i >= total) return;
  int o = i & 15;
  int bt = i >> 4;
  int t = bt % T;
  int b = bt / T;
  float acc = bias[o];
#pragma unroll
  for (int kk = 0; kk < 4; ++kk) {
    int tt = t - 1 + kk;
    if (tt >= 0 && tt < T) acc += X[(size_t)b * T + tt] * w[kk * 16 + o];
  }
  out[i] = (_Float16)acc;
}

// ---------------------------------------------------------------------------
// Pack conv output (B,T,16) linear f16 -> A-fragment layout
// [btile(4)][t][kchunk=1][lane(32)][16], K padded 16->32 with zeros.
// ---------------------------------------------------------------------------
__global__ void pack_x_k(const _Float16* __restrict__ lin,
                         _Float16* __restrict__ dst, int T) {
  int tid = blockIdx.x * blockDim.x + threadIdx.x;
  int total = 4 * T * 32;
  if (tid >= total) return;
  int lane = tid & 31;
  int rest = tid >> 5;
  int t = rest % T;
  int btile = rest / T;
  int row = lane & 15;
  const _Float16* src = lin + ((size_t)(btile * 16 + row) * T + t) * 16;
#pragma unroll
  for (int e = 0; e < 16; ++e) {
    int k = amap_k(lane, e);
    dst[(size_t)tid * 16 + e] = (k < 16) ? src[k] : (_Float16)0.f;
  }
}

// ---------------------------------------------------------------------------
// Repack f32 weight (K x N, row major) -> B-fragment layout
// [kchunk][ntile(N/16)][lane(32)][16]; optional row reversal (feature flip).
// ---------------------------------------------------------------------------
__global__ void repack_b_k(const float* __restrict__ src,
                           _Float16* __restrict__ dst, int K, int N, int rev) {
  int Kc = (K + 31) >> 5;
  int tid = blockIdx.x * blockDim.x + threadIdx.x;
  int total = Kc * (N >> 4) * 32;
  if (tid >= total) return;
  int lane = tid & 31;
  int fragi = tid >> 5;
  int ntile = fragi % (N >> 4);
  int kchunk = fragi / (N >> 4);
  int col = ntile * 16 + (lane & 15);
  int khi = (lane >> 4) ? 8 : 0;
#pragma unroll
  for (int e = 0; e < 16; ++e) {
    int k = kchunk * 32 + ((e < 8) ? e : (e + 8)) + khi;
    float v = 0.f;
    if (k < K) v = src[(size_t)(rev ? (K - 1 - k) : k) * N + col];
    dst[(size_t)tid * 16 + e] = (_Float16)v;
  }
}

// ---------------------------------------------------------------------------
// Persistent LSTM layer pair. grid=(4 batch tiles, 2 directions). One block
// loops over all T steps. For the small layers (DINC==1, H=128) all weight
// fragments are hoisted into registers (loop-invariant, 160 VGPRs/lane);
// for the big layers they stream from L2 (Wh3 = 512KB > 320KB LDS).
// ---------------------------------------------------------------------------
template <int DINC, int H>
__global__ __launch_bounds__((H / 16) * 32)
void lstm_layer_k(const v16h* __restrict__ Xfrag,    // [btile][T][DINC][32]
                  const v16h* __restrict__ Wif0, const v16h* __restrict__ Whf0,
                  const float* __restrict__ bias0,
                  const v16h* __restrict__ Wif1, const v16h* __restrict__ Whf1,
                  const float* __restrict__ bias1,
                  _Float16* __restrict__ Hout,       // [btile][T][dkcOut][32][16]
                  int dkcOut, int T) {
  constexpr int HC = H / 32;       // K chunks of recurrent GEMM
  constexpr int JT = H / 16;       // j tiles per gate == waves per block
  constexpr int NT = 4 * JT;       // n tiles across all 4 gates
  __shared__ v16h hs[HC * 32];     // h state, fragment layout
  const int tid = threadIdx.x, lane = tid & 31, wid = tid >> 5;
  const int hi = lane >> 4;
  const int jt = wid;
  const int jc = jt * 16 + (lane & 15);
  const int btile = blockIdx.x;
  const int sel = blockIdx.y;  // 0: forward weights, 1: feature-flipped set
  const v16h* __restrict__ Wif = sel ? Wif1 : Wif0;
  const v16h* __restrict__ Whf = sel ? Whf1 : Whf0;
  const float* __restrict__ bias = sel ? bias1 : bias0;
  const int hoff = sel * H;

  for (int i = tid; i < HC * 32; i += JT * 32) hs[i] = hzero();
  float creg[8];
#pragma unroll
  for (int r = 0; r < 8; ++r) creg[r] = 0.f;
  const float bi = bias[jc];
  const float bf = bias[H + jc];
  const float bg = bias[2 * H + jc];
  const float bo = bias[3 * H + jc];

  // inverse A-map constants for writing h at (row=M, k=jc)
  const int kp = jc & 31;
  const int khalf = (kp >> 3) & 1;
  const int kb2 = kp - 8 * khalf;
  const int eo = (kb2 < 8) ? kb2 : (kb2 - 8);
  const int kcl = jc >> 5;              // LDS kchunk
  const int kcg = (hoff + jc) >> 5;     // global kchunk (hoff multiple of 32)
  _Float16* hsF = (_Float16*)hs;
  __syncthreads();

  // Gate nonlinearity + state update + fragment-layout writeback.
  auto gates_out = [&](int t, v8f zi, v8f zf, v8f zg, v8f zo) {
    const size_t obase = (((size_t)btile * T + t) * dkcOut + kcg) * 32;
#pragma unroll
    for (int r = 0; r < 8; ++r) {
      float iv = sigf(zi[r] + bi);
      float fv = sigf(zf[r] + bf);
      float gv = tanh_fast(zg[r] + bg);
      float ov = sigf(zo[r] + bo);
      float c = fv * creg[r] + iv * gv;
      creg[r] = c;
      float h = ov * tanh_fast(c);
      int M = r + hi * 8;               // C/D layout row
      int lo = M + 16 * khalf;          // fragment lane for (M, jc)
      hsF[(kcl * 32 + lo) * 16 + eo] = (_Float16)h;
      Hout[(obase + lo) * 16 + eo] = (_Float16)h;
    }
  };

  if (DINC == 1) {
    // ---- small layers: weights register-resident (loop invariant) ----
    v16h wiR[4];
    v16h whR[HC][4];
#pragma unroll
    for (int g = 0; g < 4; ++g)
      wiR[g] = Wif[(size_t)(g * JT + jt) * 32 + lane];
#pragma unroll
    for (int kc = 0; kc < HC; ++kc)
#pragma unroll
      for (int g = 0; g < 4; ++g)
        whR[kc][g] = Whf[(size_t)(kc * NT + g * JT + jt) * 32 + lane];

    for (int t = 0; t < T; ++t) {
      const size_t abase = ((size_t)btile * T + t) * 32 + lane;
      if (t + 1 < T)
        __builtin_prefetch((const void*)(Xfrag + abase + 32), 0, 2);
      v16h a = Xfrag[abase];
      v8f zi = wmma16(a, wiR[0], vzero());
      v8f zf = wmma16(a, wiR[1], vzero());
      v8f zg = wmma16(a, wiR[2], vzero());
      v8f zo = wmma16(a, wiR[3], vzero());
#pragma unroll
      for (int kc = 0; kc < HC; ++kc) {
        v16h ah = hs[kc * 32 + lane];
        zi = wmma16(ah, whR[kc][0], zi);
        zf = wmma16(ah, whR[kc][1], zf);
        zg = wmma16(ah, whR[kc][2], zg);
        zo = wmma16(ah, whR[kc][3], zo);
      }
      __syncthreads();
      gates_out(t, zi, zf, zg, zo);
      __syncthreads();
    }
  } else {
    // ---- big layers: weights stream from global (L2-resident) ----
    for (int t = 0; t < T; ++t) {
      const size_t abase = (((size_t)btile * T + t) * DINC) * 32 + lane;
      if (t + 1 < T)
        __builtin_prefetch((const void*)(Xfrag + abase + (size_t)DINC * 32), 0, 2);
      v8f zi = vzero(), zf = vzero(), zg = vzero(), zo = vzero();
#pragma unroll
      for (int kc = 0; kc < DINC; ++kc) {
        v16h a = Xfrag[abase + (size_t)kc * 32];
        zi = wmma16(a, Wif[(size_t)(kc * NT + 0 * JT + jt) * 32 + lane], zi);
        zf = wmma16(a, Wif[(size_t)(kc * NT + 1 * JT + jt) * 32 + lane], zf);
        zg = wmma16(a, Wif[(size_t)(kc * NT + 2 * JT + jt) * 32 + lane], zg);
        zo = wmma16(a, Wif[(size_t)(kc * NT + 3 * JT + jt) * 32 + lane], zo);
      }
#pragma unroll
      for (int kc = 0; kc < HC; ++kc) {
        v16h a = hs[kc * 32 + lane];
        zi = wmma16(a, Whf[(size_t)(kc * NT + 0 * JT + jt) * 32 + lane], zi);
        zf = wmma16(a, Whf[(size_t)(kc * NT + 1 * JT + jt) * 32 + lane], zf);
        zg = wmma16(a, Whf[(size_t)(kc * NT + 2 * JT + jt) * 32 + lane], zg);
        zo = wmma16(a, Whf[(size_t)(kc * NT + 3 * JT + jt) * 32 + lane], zo);
      }
      __syncthreads();
      gates_out(t, zi, zf, zg, zo);
      __syncthreads();
    }
  }
}

// ---------------------------------------------------------------------------
// Attention score. One block per (btile, t): rows = batch btile*16..+15.
// ---------------------------------------------------------------------------
__global__ __launch_bounds__(128)
void att_score_k(const v16h* __restrict__ encF, const v16h* __restrict__ awF,
                 const float* __restrict__ attWb, const float* __restrict__ attV,
                 const float* __restrict__ attVb, float* __restrict__ score,
                 int T) {
  __shared__ float ssc[16];
  const int tid = threadIdx.x, lane = tid & 31, wid = tid >> 5;
  const int hi = lane >> 4;
  const int btile = blockIdx.x / T;
  const int t = blockIdx.x % T;
  if (tid < 16) ssc[tid] = 0.f;
  __syncthreads();
  const size_t abase = (((size_t)btile * T + t) * 16) * 32 + lane;
#pragma unroll
  for (int tile = 0; tile < 2; ++tile) {
    const int ntile = wid * 2 + tile;
    v8f acc = vzero();
#pragma unroll
    for (int kc = 0; kc < 16; ++kc)
      acc = wmma16(encF[abase + (size_t)kc * 32],
                   awF[(size_t)(kc * 8 + ntile) * 32 + lane], acc);
    const int col = ntile * 16 + (lane & 15);
    const float bcol = attWb[col], vcol = attV[col];
#pragma unroll
    for (int r = 0; r < 8; ++r)
      atomicAdd(&ssc[r + hi * 8], tanh_fast(acc[r] + bcol) * vcol);
  }
  __syncthreads();
  if (tid < 16)
    score[(size_t)(btile * 16 + tid) * T + t] = ssc[tid] + attVb[0];
}

// ---------------------------------------------------------------------------
// Softmax over T + context. Reads enc from fragment layout via per-thread
// constant offsets (stride 8192 elements per t).
// ---------------------------------------------------------------------------
__global__ __launch_bounds__(256)
void softmax_ctx_k(const float* __restrict__ score,
                   const _Float16* __restrict__ encF, float* __restrict__ ctx,
                   int T) {
  __shared__ float red[256];
  const int b = blockIdx.x, tid = threadIdx.x;
  const int btile = b >> 4, row = b & 15;
  const float* sb = score + (size_t)b * T;
  float m = -1e30f;
  for (int t = tid; t < T; t += 256) m = fmaxf(m, sb[t]);
  red[tid] = m;
  __syncthreads();
  for (int s = 128; s > 0; s >>= 1) {
    if (tid < s) red[tid] = fmaxf(red[tid], red[tid + s]);
    __syncthreads();
  }
  const float mx = red[0];
  __syncthreads();
  float sum = 0.f;
  for (int t = tid; t < T; t += 256) sum += __expf(sb[t] - mx);
  red[tid] = sum;
  __syncthreads();
  for (int s = 128; s > 0; s >>= 1) {
    if (tid < s) red[tid] += red[tid + s];
    __syncthreads();
  }
  const float inv = 1.f / red[0];

  auto coff = [&](int c) -> size_t {  // fragment offset of channel c for row b
    int kchunk = c >> 5;
    int kp = c & 31;
    int khalf = (kp >> 3) & 1;
    int lane = row + 16 * khalf;
    int kb = kp - 8 * khalf;
    int e = (kb < 8) ? kb : kb - 8;
    return (size_t)(kchunk * 32 + lane) * 16 + e;
  };
  const size_t o0 = coff(tid), o1 = coff(tid + 256);
  float a0 = 0.f, a1 = 0.f;
  for (int t = 0; t < T; ++t) {
    const float w = __expf(sb[t] - mx) * inv;
    const size_t base = ((size_t)btile * T + t) * 8192;  // 16*32*16 elem per t
    a0 += w * (float)encF[base + o0];
    a1 += w * (float)encF[base + o1];
  }
  ctx[(size_t)b * 512 + tid] = a0;
  ctx[(size_t)b * 512 + tid + 256] = a1;
}

// ---------------------------------------------------------------------------
// Head: out[b] = tanh(ctx @ d1W + d1b) @ d2W + d2b   (tiny)
// ---------------------------------------------------------------------------
__global__ __launch_bounds__(128)
void head_k(const float* __restrict__ ctx, const float* __restrict__ d1W,
            const float* __restrict__ d1b, const float* __restrict__ d2W,
            const float* __restrict__ d2b, float* __restrict__ out) {
  __shared__ float red[128];
  const int b = blockIdx.x, j = threadIdx.x;
  float acc = d1b[j];
  for (int k = 0; k < 512; ++k) acc += ctx[(size_t)b * 512 + k] * d1W[k * 128 + j];
  red[j] = tanh_fast(acc) * d2W[j];
  __syncthreads();
  for (int s = 64; s > 0; s >>= 1) {
    if (j < s) red[j] += red[j + s];
    __syncthreads();
  }
  if (j == 0) out[b] = red[0] + d2b[0];
}

// ---------------------------------------------------------------------------
// Host launcher
// ---------------------------------------------------------------------------
extern "C" void kernel_launch(void* const* d_in, const int* in_sizes, int n_in,
                              void* d_out, int out_size, void* d_ws,
                              size_t ws_size, hipStream_t stream) {
  (void)in_sizes; (void)n_in; (void)out_size; (void)ws_size;
  const int Bv = 64, T = 1024;
  const int BT = Bv * T;

  const float* X      = (const float*)d_in[0];
  const float* conv_w = (const float*)d_in[1];
  const float* conv_b = (const float*)d_in[2];
  const float* Wi1 = (const float*)d_in[3];
  const float* Wh1 = (const float*)d_in[4];
  const float* b1  = (const float*)d_in[5];
  const float* Wi2 = (const float*)d_in[6];
  const float* Wh2 = (const float*)d_in[7];
  const float* b2  = (const float*)d_in[8];
  const float* Wi3 = (const float*)d_in[9];
  const float* Wh3 = (const float*)d_in[10];
  const float* b3  = (const float*)d_in[11];
  const float* Wi4 = (const float*)d_in[12];
  const float* Wh4 = (const float*)d_in[13];
  const float* b4  = (const float*)d_in[14];
  const float* attW  = (const float*)d_in[15];
  const float* attWb = (const float*)d_in[16];
  const float* attV  = (const float*)d_in[17];
  const float* attVb = (const float*)d_in[18];
  const float* d1W = (const float*)d_in[19];
  const float* d1b = (const float*)d_in[20];
  const float* d2W = (const float*)d_in[21];
  const float* d2b = (const float*)d_in[22];

  char* ws = (char*)d_ws;
  size_t off = 0;
  auto take = [&](size_t bytes) {
    char* p = ws + off;
    off = (off + bytes + 255) & ~(size_t)255;
    return p;
  };
  // activations
  _Float16* xlin  = (_Float16*)take((size_t)BT * 16 * 2);          // (B,T,16)
  _Float16* xF    = (_Float16*)take((size_t)4 * T * 1 * 512 * 2);  // frags
  _Float16* c12F  = (_Float16*)take((size_t)4 * T * 8 * 512 * 2);  // D=256
  _Float16* encF  = (_Float16*)take((size_t)4 * T * 16 * 512 * 2); // D=512
  float*    score = (float*)take((size_t)BT * 4);
  float*    ctx   = (float*)take((size_t)Bv * 512 * 4);
  // weight fragments
  _Float16* wi1F = (_Float16*)take((size_t)1 * 32 * 512 * 2);
  _Float16* wh1F = (_Float16*)take((size_t)4 * 32 * 512 * 2);
  _Float16* wi2F = (_Float16*)take((size_t)1 * 32 * 512 * 2);
  _Float16* wh2F = (_Float16*)take((size_t)4 * 32 * 512 * 2);
  _Float16* wi3F = (_Float16*)take((size_t)8 * 64 * 512 * 2);
  _Float16* wh3F = (_Float16*)take((size_t)8 * 64 * 512 * 2);
  _Float16* wi4F = (_Float16*)take((size_t)8 * 64 * 512 * 2);
  _Float16* wh4F = (_Float16*)take((size_t)8 * 64 * 512 * 2);
  _Float16* awF  = (_Float16*)take((size_t)16 * 8 * 512 * 2);

  auto repack = [&](const float* s, _Float16* d, int K, int N, int rev) {
    int total = ((K + 31) >> 5) * (N >> 4) * 32;
    repack_b_k<<<(total + 255) / 256, 256, 0, stream>>>(s, d, K, N, rev);
  };
  repack(Wi1, wi1F, 16, 512, 0);
  repack(Wh1, wh1F, 128, 512, 0);
  repack(Wi2, wi2F, 16, 512, 1);   // feature flip folded into row reversal
  repack(Wh2, wh2F, 128, 512, 0);
  repack(Wi3, wi3F, 256, 1024, 0);
  repack(Wh3, wh3F, 256, 1024, 0);
  repack(Wi4, wi4F, 256, 1024, 1);
  repack(Wh4, wh4F, 256, 1024, 0);
  repack(attW, awF, 512, 128, 0);

  conv_k<<<(BT * 16 + 255) / 256, 256, 0, stream>>>(X, conv_w, conv_b, xlin, T,
                                                    BT * 16);
  pack_x_k<<<(4 * T * 32 + 255) / 256, 256, 0, stream>>>(xlin, xF, T);

  // Layers 1 & 2 concurrently: grid (4 batch tiles, 2 directions)
  lstm_layer_k<1, 128><<<dim3(4, 2), 256, 0, stream>>>(
      (const v16h*)xF, (const v16h*)wi1F, (const v16h*)wh1F, b1,
      (const v16h*)wi2F, (const v16h*)wh2F, b2, c12F, 8, T);
  // Layers 3 & 4 concurrently
  lstm_layer_k<8, 256><<<dim3(4, 2), 512, 0, stream>>>(
      (const v16h*)c12F, (const v16h*)wi3F, (const v16h*)wh3F, b3,
      (const v16h*)wi4F, (const v16h*)wh4F, b4, encF, 16, T);

  att_score_k<<<4 * T, 128, 0, stream>>>((const v16h*)encF, (const v16h*)awF,
                                         attWb, attV, attVb, score, T);
  softmax_ctx_k<<<Bv, 256, 0, stream>>>(score, encF, ctx, T);
  head_k<<<Bv, 128, 0, stream>>>(ctx, d1W, d1b, d2W, d2b, (float*)d_out);
}
